// SelfAttention_33852932227207
// MI455X (gfx1250) — matrix operand order repaired and checked by reading, not compiled, not run on hardware
//
#include <hip/hip_runtime.h>

typedef _Float16 v8h  __attribute__((ext_vector_type(8)));
typedef _Float16 v16h __attribute__((ext_vector_type(16)));
typedef float    v8f  __attribute__((ext_vector_type(8)));

#define B_     4
#define S_     2048
#define DIN_   1024
#define D_     512
#define BS_    (B_ * S_)        // 8192
#define INV_TEMP 0.04419417382415922f  // 1/sqrt(512)

__device__ __forceinline__ v16h cat16(v8h lo, v8h hi) {
    v16h r;
#pragma unroll
    for (int i = 0; i < 8; ++i) { r[i] = lo[i]; r[i + 8] = hi[i]; }
    return r;
}

// A-fragment (16x32 f16, MxK): lane%16 = row M, group g = lane/16.
// elements 0..15 -> K = kk + 16g + [0,16)  (16 contiguous halves per lane)
__device__ __forceinline__ v16h load_afrag(const _Float16* rowPtr, int kk, int g) {
    v8h lo = *(const v8h*)(rowPtr + kk + 16 * g);
    v8h hi = *(const v8h*)(rowPtr + kk + 16 * g + 8);
    return cat16(lo, hi);
}

// B-fragment (32x16 f16, KxN): lane%16 = column N, group g = lane/16.
// elements 0..15 -> K = 16g + [0,16)  (16 contiguous halves per lane)
__device__ __forceinline__ v16h load_bfrag16(const _Float16* p) {
    v8h lo = *(const v8h*)(p);
    v8h hi = *(const v8h*)(p + 8);
    return cat16(lo, hi);
}

__global__ void __launch_bounds__(256) cvt_f32_f16(const float* __restrict__ src,
                                                   _Float16* __restrict__ dst, int n) {
    int i = blockIdx.x * blockDim.x + threadIdx.x;
    int stride = gridDim.x * blockDim.x;
    for (; i < n; i += stride) dst[i] = (_Float16)src[i];
}

// C[m][n] = sum_k A[m][k] * W[n][k]   (A: [M,K] f16 row-major, W: [N,K] f16 row-major)
// TOUT=false: store f16 row-major C [M,N] (ldo = N)
// TOUT=true : store f16 transposed  C [N,M] (ldo = M), contiguous per-lane stores
template <bool TOUT>
__global__ void __launch_bounds__(256, 1)
gemm16(const _Float16* __restrict__ A, const _Float16* __restrict__ W,
       _Float16* __restrict__ Cout, int M, int N, int K, int ldo) {
    const int tid  = threadIdx.x;
    const int lane = tid & 31, g = lane >> 4, l = lane & 15;
    const int wid  = blockIdx.x * (blockDim.x >> 5) + (tid >> 5);
    const int ntile4 = N >> 6;
    const int tm = wid / ntile4, tn = wid % ntile4;
    if (tm * 16 >= M) return;
    const int m0 = tm * 16, n0 = tn * 64;

    const _Float16* arow = A + (size_t)(m0 + l) * K;
    v8f acc[4] = {};

#pragma unroll 4
    for (int kk = 0; kk < K; kk += 32) {
        v16h a = load_afrag(arow, kk, g);
#pragma unroll
        for (int j = 0; j < 4; ++j) {
            v16h b = load_bfrag16(W + (size_t)(n0 + 16 * j + l) * K + kk + 16 * g);
            acc[j] = __builtin_amdgcn_wmma_f32_16x16x32_f16(
                false, a, false, b, (short)0, acc[j], false, false);
        }
    }

#pragma unroll
    for (int j = 0; j < 4; ++j) {
        if (TOUT) {
            v8h h;
#pragma unroll
            for (int r = 0; r < 8; ++r) h[r] = (_Float16)acc[j][r];
            *(v8h*)(Cout + (size_t)(n0 + 16 * j + l) * ldo + m0 + 8 * g) = h;
        } else {
#pragma unroll
            for (int r = 0; r < 8; ++r)
                Cout[(size_t)(m0 + r + 8 * g) * ldo + (n0 + 16 * j + l)] = (_Float16)acc[j][r];
        }
    }
}

// One workgroup (256 thr = 8 waves) per (batch, 16-query block).
// LDS: scores f32 [16][S] (128KB) | alpha f16 [16][S] (64KB) | stats (1152B)
#define ATTN_SMEM (16 * S_ * 4 + 16 * S_ * 2 + 288 * 4)

__global__ void __launch_bounds__(256, 1)
attn_kernel(const _Float16* __restrict__ qh, const _Float16* __restrict__ kah,
            const _Float16* __restrict__ uT, float* __restrict__ out) {
    extern __shared__ char smem[];
    float*    sc    = (float*)smem;                               // [16][S]
    _Float16* al    = (_Float16*)(smem + 16 * S_ * 4);            // [16][S]
    float*    stats = (float*)(smem + 16 * S_ * 4 + 16 * S_ * 2); // [288]

    const int tid  = threadIdx.x;
    const int lane = tid & 31, g = lane >> 4, l = lane & 15, wid = tid >> 5;
    const int batch = blockIdx.x >> 7;       // S/16 = 128 query tiles per batch
    const int qt    = blockIdx.x & 127;
    const int qbase = batch * S_ + qt * 16;

    // ---- Phase 1: scores e = q @ ka^T / temp, staged to LDS ----
    // Explicitly hoist the 16 loop-invariant q A-fragments (128 VGPRs; fits with
    // launch_bounds since LDS already limits this kernel to 1 workgroup/WGP).
    const _Float16* qrow = qh + (size_t)(qbase + l) * D_;
    v16h qfrag[D_ / 32];
#pragma unroll
    for (int ks = 0; ks < D_ / 32; ++ks) qfrag[ks] = load_afrag(qrow, ks * 32, g);

#pragma unroll 1
    for (int t = wid; t < S_ / 16; t += 8) {
        v8f c = {};
        const _Float16* karow = kah + (size_t)(batch * S_ + t * 16 + l) * D_;
#pragma unroll
        for (int ks = 0; ks < D_ / 32; ++ks) {
            v16h b = load_bfrag16(karow + ks * 32 + 16 * g);
            c = __builtin_amdgcn_wmma_f32_16x16x32_f16(false, qfrag[ks], false, b,
                                                       (short)0, c, false, false);
        }
#pragma unroll
        for (int r = 0; r < 8; ++r)
            sc[(r + 8 * g) * S_ + t * 16 + l] = c[r] * INV_TEMP;
    }
    __syncthreads();

    // ---- Phase 2: row-wise softmax stats, alpha = exp(s - max) as f16 ----
    const int row = tid >> 4, jj = tid & 15;
    float mx = -3.0e38f;
    for (int col = jj; col < S_; col += 16) mx = fmaxf(mx, sc[row * S_ + col]);
    stats[row * 16 + jj] = mx;
    __syncthreads();
    if (tid < 16) {
        float m = -3.0e38f;
        for (int i = 0; i < 16; ++i) m = fmaxf(m, stats[tid * 16 + i]);
        stats[256 + tid] = m;
    }
    __syncthreads();
    const float rmax = stats[256 + row];
    float sum = 0.f;
    for (int col = jj; col < S_; col += 16) {
        float e = __expf(sc[row * S_ + col] - rmax);
        al[row * S_ + col] = (_Float16)e;
        sum += e;
    }
    stats[row * 16 + jj] = sum;
    __syncthreads();
    if (tid < 16) {
        float s = 0.f;
        for (int i = 0; i < 16; ++i) s += stats[tid * 16 + i];
        stats[272 + tid] = 1.0f / s;
    }
    __syncthreads();

    // ---- Phase 3: out = alpha @ u ; each wave owns a 64-wide DV slice ----
    const int n0 = wid * 64;
    v8f acc[4] = {};
#pragma unroll 1
    for (int t = 0; t < S_ / 16; ++t) {
        const _Float16* abase = al + l * S_ + t * 16;  // A-frag: lane = query row
        v8h alo = *(const v8h*)(abase + 16 * g);
        v8h ahi = *(const v8h*)(abase + 16 * g + 8);
        v16h a  = cat16(alo, ahi);
#pragma unroll
        for (int j = 0; j < 4; ++j) {
            const _Float16* bp =
                uT + (size_t)(n0 + 16 * j + l) * BS_ + batch * S_ + t * 16 + 16 * g;
            v16h b = load_bfrag16(bp);
            acc[j] = __builtin_amdgcn_wmma_f32_16x16x32_f16(false, a, false, b, (short)0,
                                                            acc[j], false, false);
        }
    }
#pragma unroll
    for (int j = 0; j < 4; ++j)
#pragma unroll
        for (int r = 0; r < 8; ++r) {
            int m = r + 8 * g;
            out[(size_t)(qbase + m) * D_ + n0 + 16 * j + l] = acc[j][r] * stats[272 + m];
        }
}

extern "C" void kernel_launch(void* const* d_in, const int* in_sizes, int n_in,
                              void* d_out, int out_size, void* d_ws, size_t ws_size,
                              hipStream_t stream) {
    (void)in_sizes; (void)n_in; (void)out_size; (void)ws_size;
    const float* x  = (const float*)d_in[0];
    const float* Wk = (const float*)d_in[1];
    const float* Wq = (const float*)d_in[2];
    const float* Wv = (const float*)d_in[3];
    const float* Wa = (const float*)d_in[4];
    const float* Wb = (const float*)d_in[5];

    char* ws = (char*)d_ws;
    _Float16* XH  = (_Float16*)(ws + 0);           // 16 MB  [8192,1024]
    _Float16* QH  = (_Float16*)(ws + 16777216);    //  8 MB  [8192,512]
    _Float16* KH  = (_Float16*)(ws + 25165824);    //  8 MB
    _Float16* VH  = (_Float16*)(ws + 33554432);    //  8 MB
    _Float16* KAH = (_Float16*)(ws + 41943040);    //  8 MB
    _Float16* UT  = (_Float16*)(ws + 50331648);    //  8 MB  [512,8192] transposed
    _Float16* WKH = (_Float16*)(ws + 58720256);    //  1 MB  [512,1024]
    _Float16* WQH = (_Float16*)(ws + 59768832);
    _Float16* WVH = (_Float16*)(ws + 60817408);
    _Float16* WAH = (_Float16*)(ws + 61865984);    // 512 KB [512,512]
    _Float16* WBH = (_Float16*)(ws + 62390272);

    // f32 -> f16 staging
    cvt_f32_f16<<<4096, 256, 0, stream>>>(x,  XH,  BS_ * DIN_);
    cvt_f32_f16<<<512,  256, 0, stream>>>(Wk, WKH, D_ * DIN_);
    cvt_f32_f16<<<512,  256, 0, stream>>>(Wq, WQH, D_ * DIN_);
    cvt_f32_f16<<<512,  256, 0, stream>>>(Wv, WVH, D_ * DIN_);
    cvt_f32_f16<<<256,  256, 0, stream>>>(Wa, WAH, D_ * D_);
    cvt_f32_f16<<<256,  256, 0, stream>>>(Wb, WBH, D_ * D_);

    // projections: q/k/v = x @ W.T      grid: (8192/16)*(512/64)=4096 waves -> 512 blocks
    gemm16<false><<<512, 256, 0, stream>>>(XH, WQH, QH, BS_, D_, DIN_, D_);
    gemm16<false><<<512, 256, 0, stream>>>(XH, WKH, KH, BS_, D_, DIN_, D_);
    gemm16<false><<<512, 256, 0, stream>>>(XH, WVH, VH, BS_, D_, DIN_, D_);
    // ka = k @ Wa.T ; uT = (v @ Wb.T)^T
    gemm16<false><<<512, 256, 0, stream>>>(KH, WAH, KAH, BS_, D_, D_, D_);
    gemm16<true ><<<512, 256, 0, stream>>>(VH, WBH, UT,  BS_, D_, D_, BS_);

    // attention: one block per (batch, 16-query tile) = 4*128 = 512 blocks
    hipFuncSetAttribute(reinterpret_cast<const void*>(attn_kernel),
                        hipFuncAttributeMaxDynamicSharedMemorySize, ATTN_SMEM);
    attn_kernel<<<512, 256, ATTN_SMEM, stream>>>(QH, KAH, UT, (float*)d_out);
}
